// MultiHeadSelfAttention_42365557407767
// MI455X (gfx1250) — matrix operand order repaired and checked by reading, not compiled, not run on hardware
//
#include <hip/hip_runtime.h>
#include <hip/hip_bf16.h>

typedef _Float16 half_t;
typedef __attribute__((ext_vector_type(16))) _Float16 v16h;
typedef __attribute__((ext_vector_type(8)))  _Float16 v8h;
typedef __attribute__((ext_vector_type(4)))  _Float16 v4h;
typedef __attribute__((ext_vector_type(8)))  float    v8f;
typedef __attribute__((ext_vector_type(4)))  float    v4f;
typedef __attribute__((ext_vector_type(4)))  unsigned int u32x4;
typedef __attribute__((ext_vector_type(8)))  unsigned int u32x8;

#define DEV static __device__ __forceinline__

constexpr int Bsz  = 2;
constexpr int Tseq = 2048;
constexpr int Cdim = 1024;
constexpr int Hn   = 16;
constexpr int Dh   = 64;            // head dim
constexpr int Mrows = Bsz * Tseq;   // 4096

// ---------------------------------------------------------------------------
// CDNA5 async data movement helpers
// ---------------------------------------------------------------------------

// Per-lane async copy: 16 contiguous bytes global -> 16 contiguous bytes LDS.
// Tracked on ASYNCcnt.  (CDNA5 ISA opcode 98 / CDNA5 ISA)
DEV void async_load_b128(const half_t* g, half_t* l) {
    unsigned lds_addr = (unsigned)(unsigned long long)(uintptr_t)l; // LDS offset = addr[31:0]
    unsigned long long gaddr = (unsigned long long)(uintptr_t)g;
    asm volatile("global_load_async_to_lds_b128 %0, %1, off"
                 :: "v"(lds_addr), "v"(gaddr) : "memory");
}

DEV void wait_async0() {
    asm volatile("s_wait_asynccnt 0x0" ::: "memory");
}

DEV void wait_tensor0() {
#if __has_builtin(__builtin_amdgcn_s_wait_tensorcnt)
    __builtin_amdgcn_s_wait_tensorcnt(0);
#else
    asm volatile("s_wait_tensorcnt 0x0" ::: "memory");
#endif
}

// TDM: 2D tile of 16-bit elements, global -> LDS.  Wave-level DMA, TENSORcnt.
// Descriptor layout per CDNA5 ISA (D# groups 0 and 1,
// 2-group form: VADDR2/3 = NULL => tensors up to 2D).
//   tile_d0 = elements per row, tile_d1 = rows, stride0 = row stride (elems)
//   pad_*  = optional LDS padding (pad_interval code: 0=2dw..7=256dw;
//            pad_amount code: N => N+1 dwords)
DEV void tdm_load_2d_f16(const half_t* gsrc, const half_t* ldst,
                         unsigned tile_d0, unsigned tile_d1,
                         unsigned long long stride0,
                         unsigned pad_en, unsigned pad_interval,
                         unsigned pad_amount)
{
    unsigned long long ga = (unsigned long long)(uintptr_t)gsrc;
    unsigned lds = (unsigned)(unsigned long long)(uintptr_t)ldst;
    u32x4 g0;
    g0[0] = 1u;                                         // count=1, user mode
    g0[1] = lds;                                        // lds_addr
    g0[2] = (unsigned)(ga & 0xFFFFFFFFu);               // global_addr[31:0]
    g0[3] = (unsigned)((ga >> 32) & 0x01FFFFFFu)        // global_addr[56:32]
          | (2u << 30);                                 // type = 2 ("image")
    u32x8 g1;
    g1[0] = (1u << 16)                                  // data_size = 1 (2B)
          | (pad_en << 20) | (pad_interval << 22) | (pad_amount << 25);
    g1[1] = (tile_d0 & 0xFFFFu) << 16;                  // tensor_dim0.lo
    g1[2] = (tile_d0 >> 16) | ((tile_d1 & 0xFFFFu) << 16);  // d0.hi | tensor_dim1.lo
    g1[3] = (tile_d1 >> 16) | ((tile_d0 & 0xFFFFu) << 16);  // d1.hi | tile_dim0
    g1[4] = (tile_d1 & 0xFFFFu);                        // tile_dim1 | tile_dim2=0
    g1[5] = (unsigned)(stride0 & 0xFFFFFFFFull);        // tensor_dim0_stride.lo
    g1[6] = (unsigned)(stride0 >> 32);                  // .hi | dim1_stride.lo=0
    g1[7] = 0u;                                         // dim1_stride.hi
    asm volatile("tensor_load_to_lds %0, %1" :: "s"(g0), "s"(g1) : "memory");
}

// ---------------------------------------------------------------------------
// WMMA fragment loaders (lane layouts per CDNA5 ISA 7.12.2, wave32)
// ---------------------------------------------------------------------------
DEV v16h load_frag_a(const half_t* base, int ldk, int lane) {
    int r  = lane & 15;
    int kb = (lane < 16) ? 0 : 8;
    const half_t* p = base + r * ldk + kb;
    v8h lo = *(const v8h*)(p);
    v8h hi = *(const v8h*)(p + 16);
    v16h out;
#pragma unroll
    for (int e = 0; e < 8; ++e) { out[e] = lo[e]; out[e + 8] = hi[e]; }
    return out;
}

DEV v16h load_frag_b(const half_t* base, int ldk, int lane) {
    int c  = lane & 15;
    int kb = (lane < 16) ? 0 : 8;
    const half_t* p = base + c * ldk + kb;
    v8h lo = *(const v8h*)(p);
    v8h hi = *(const v8h*)(p + 16);
    v16h out;
#pragma unroll
    for (int e = 0; e < 8; ++e) { out[e] = lo[e]; out[e + 8] = hi[e]; }
    return out;
}

DEV v8f wmma_f16(v16h a, v16h b, v8f c) {
    return __builtin_amdgcn_wmma_f32_16x16x32_f16(false, a, false, b,
                                                  (short)0, c, false, false);
}

// ---------------------------------------------------------------------------
// fp32 -> f16 conversion (vectorized x4)
// ---------------------------------------------------------------------------
__global__ __launch_bounds__(256) void cvt_f32_f16(const float* __restrict__ in,
                                                   half_t* __restrict__ out, int n4) {
    int i = blockIdx.x * blockDim.x + threadIdx.x;
    if (i < n4) {
        v4f x = *(const v4f*)(in + (size_t)i * 4);
        v4h y;
#pragma unroll
        for (int j = 0; j < 4; ++j) y[j] = (half_t)x[j];
        *(v4h*)(out + (size_t)i * 4) = y;
    }
}

// ---------------------------------------------------------------------------
// WMMA GEMM: C[M,N] = A[M,K] * B[K,N], f16 in / f32 accumulate.
// 128x128x32 tile, 256 threads (8 waves).  A tiles staged by TDM (wave 0,
// overlapped with compute, TENSORcnt); B tiles reg-staged + transposed.
// MODE 0: scatter qkv columns into Q/K/V [B,H,T,D] f16.   MODE 1: bias+f32.
// ---------------------------------------------------------------------------
template <int MODE>
__global__ __launch_bounds__(256) void gemm_wmma(
    const half_t* __restrict__ A, const half_t* __restrict__ Bm,
    int K, int N,
    half_t* __restrict__ Qh, half_t* __restrict__ Kh, half_t* __restrict__ Vh,
    float* __restrict__ outp, const float* __restrict__ bias)
{
    constexpr int BM = 128, BN = 128, BK = 32;
    constexpr int LDB = BK + 8;           // padded [n][k] stride (halfs)

    __shared__ half_t sA[2][BM * BK];     // [m][k], 8KB each (rows = 64B, no pad)
    __shared__ half_t sB[2][BN * LDB];    // [n][k] transposed, 10KB each

    const int tid  = threadIdx.x;
    const int lane = tid & 31;
    const int wave = tid >> 5;
    const int wm   = wave >> 2;           // 0..1
    const int wn   = wave & 3;            // 0..3
    const int m0   = blockIdx.y * BM;
    const int n0   = blockIdx.x * BN;

    v8h bReg[2];

    auto loadB = [&](int kt) {
#pragma unroll
        for (int i = 0; i < 2; ++i) {
            int c = tid + i * 256;
            int k = c >> 4, nn = (c & 15) * 8;
            bReg[i] = *(const v8h*)(Bm + (size_t)(kt * BK + k) * N + n0 + nn);
        }
    };
    auto storeB = [&](int buf) {
#pragma unroll
        for (int i = 0; i < 2; ++i) {
            int c = tid + i * 256;
            int k = c >> 4, nn = (c & 15) * 8;
#pragma unroll
            for (int j = 0; j < 8; ++j)              // transpose into [n][k]
                sB[buf][(nn + j) * LDB + k] = bReg[i][j];
        }
    };
    auto tdmA = [&](int kt, int buf) {               // whole 128x32 tile via TDM
        tdm_load_2d_f16(A + (size_t)m0 * K + kt * BK, &sA[buf][0],
                        BK, BM, (unsigned long long)K, 0, 0, 0);
    };

    v8f acc[4][2];
#pragma unroll
    for (int i = 0; i < 4; ++i)
#pragma unroll
        for (int j = 0; j < 2; ++j) acc[i][j] = (v8f)0.0f;

    const int NK = K / BK;
    if (wave == 0) tdmA(0, 0);
    loadB(0);
    storeB(0);
    if (wave == 0) wait_tensor0();
    __syncthreads();

    for (int kt = 0; kt < NK; ++kt) {
        const int cur = kt & 1;
        if (kt + 1 < NK) {
            loadB(kt + 1);
            if (wave == 0) tdmA(kt + 1, cur ^ 1);    // DMA next A tile now
        }
        if (kt + 2 < NK)
            __builtin_prefetch(Bm + (size_t)((kt + 2) * BK + (tid >> 3)) * N + n0, 0, 1);

        v16h bf[2];
#pragma unroll
        for (int j = 0; j < 2; ++j)
            bf[j] = load_frag_b(&sB[cur][(wn * 32 + j * 16) * LDB], LDB, lane);
#pragma unroll
        for (int i = 0; i < 4; ++i) {
            v16h af = load_frag_a(&sA[cur][(wm * 64 + i * 16) * BK], BK, lane);
#pragma unroll
            for (int j = 0; j < 2; ++j)
                acc[i][j] = wmma_f16(af, bf[j], acc[i][j]);
        }

        if (kt + 1 < NK) {
            storeB(cur ^ 1);
            if (wave == 0) wait_tensor0();           // A(kt+1) resident pre-barrier
        }
        __syncthreads();
    }

    // ---------------- epilogue ----------------
    const int rowHi = (lane >> 4) << 3;              // +8 for lanes 16..31
#pragma unroll
    for (int i = 0; i < 4; ++i) {
#pragma unroll
        for (int j = 0; j < 2; ++j) {
            int mbase = m0 + wm * 64 + i * 16 + rowHi;
            int n = n0 + wn * 32 + j * 16 + (lane & 15);
            if (MODE == 1) {
                float bv = bias[n];
#pragma unroll
                for (int r = 0; r < 8; ++r)
                    outp[(size_t)(mbase + r) * N + n] = acc[i][j][r] + bv;
            } else {
                int which = n >> 10;                 // 0:q 1:k 2:v
                int c = n & (Cdim - 1);
                int h = c >> 6, d = c & (Dh - 1);
                half_t* dst = (which == 0) ? Qh : (which == 1) ? Kh : Vh;
#pragma unroll
                for (int r = 0; r < 8; ++r) {
                    int m = mbase + r;
                    int b = m >> 11, t = m & (Tseq - 1);
                    dst[(((size_t)(b * Hn + h)) * Tseq + t) * Dh + d] =
                        (half_t)acc[i][j][r];
                }
            }
        }
    }
}

// ---------------------------------------------------------------------------
// Flash attention (causal), f16 QKV [B,H,T,D], online softmax, f32 accum.
// K tiles staged via global_load_async_to_lds_b128 (ASYNCcnt); V staged with
// register transpose; P converted C->A layout via wave-private LDS scratch.
// ---------------------------------------------------------------------------
__global__ __launch_bounds__(256) void attn_flash(
    const half_t* __restrict__ Qg, const half_t* __restrict__ Kg,
    const half_t* __restrict__ Vg, half_t* __restrict__ Og)
{
    constexpr int QT = 128, KT = 64, LDK = Dh + 8;   // 72-half padded stride
    __shared__ half_t sK[KT * LDK];                  // [key][d]
    __shared__ half_t sV[Dh * LDK];                  // [d][key]
    __shared__ half_t sP[8 * 16 * LDK];              // per-wave P scratch

    const int tid  = threadIdx.x;
    const int lane = tid & 31;
    const int wave = tid >> 5;
    const int bh   = blockIdx.y;                     // 0..B*H-1
    const int q0   = blockIdx.x * QT;
    const int qw   = q0 + wave * 16;                 // wave's query base

    const half_t* Qb = Qg + (size_t)bh * Tseq * Dh;
    const half_t* Kb = Kg + (size_t)bh * Tseq * Dh;
    const half_t* Vb = Vg + (size_t)bh * Tseq * Dh;

    const float NEG_INF = -__builtin_inff();

    // Q fragments: native A-layout straight from global (2 k-steps over D=64)
    v16h qf[2];
    {
        int r  = lane & 15;
        int kb = (lane < 16) ? 0 : 8;
        const half_t* p = Qb + (size_t)(qw + r) * Dh + kb;
#pragma unroll
        for (int s = 0; s < 2; ++s) {
            v8h lo = *(const v8h*)(p + s * 32);
            v8h hi = *(const v8h*)(p + s * 32 + 16);
#pragma unroll
            for (int e = 0; e < 8; ++e) { qf[s][e] = lo[e]; qf[s][e + 8] = hi[e]; }
        }
    }

    v8f oacc[4];
#pragma unroll
    for (int dt = 0; dt < 4; ++dt) oacc[dt] = (v8f)0.0f;
    float mrow[8], lrow[8];
#pragma unroll
    for (int r = 0; r < 8; ++r) { mrow[r] = NEG_INF; lrow[r] = 0.0f; }

    const int rowHi = (lane >> 4) << 3;
    const int qrow0 = qw + rowHi;
    const int nkt = (q0 + QT - 1) / KT + 1;          // key tiles to visit

    for (int kt = 0; kt < nkt; ++kt) {
        const int k0 = kt * KT;
        // stage K [key][d] via async global->LDS copies (ASYNCcnt)
#pragma unroll
        for (int i = 0; i < 2; ++i) {
            int c = tid + i * 256;                   // 512 chunks of 8 halfs
            int row = c >> 3, kc = (c & 7) * 8;
            async_load_b128(Kb + (size_t)(k0 + row) * Dh + kc,
                            &sK[row * LDK + kc]);
        }
        // stage V transposed [d][key]
#pragma unroll
        for (int i = 0; i < 2; ++i) {
            int c = tid + i * 256;
            int row = c >> 3, dc = (c & 7) * 8;
            v8h v = *(const v8h*)(Vb + (size_t)(k0 + row) * Dh + dc);
#pragma unroll
            for (int j = 0; j < 8; ++j) sV[(dc + j) * LDK + row] = v[j];
        }
        if (kt + 1 < nkt) {
            __builtin_prefetch(Kb + (size_t)(k0 + KT + (tid >> 2)) * Dh, 0, 1);
            __builtin_prefetch(Vb + (size_t)(k0 + KT + (tid >> 2)) * Dh, 0, 1);
        }
        wait_async0();                               // K tile resident
        __syncthreads();

        if (k0 <= qw + 15) {                         // wave-uniform skip
            // scores S = scale * Q K^T   (16 x 64)
            v8f sc[4];
#pragma unroll
            for (int j = 0; j < 4; ++j) {
                v8f a = (v8f)0.0f;
#pragma unroll
                for (int s = 0; s < 2; ++s) {
                    v16h bf = load_frag_b(&sK[(j * 16) * LDK + s * 32], LDK, lane);
                    a = wmma_f16(qf[s], bf, a);
                }
                sc[j] = a * 0.125f;                  // 1/sqrt(64)
            }
            // causal mask (only diagonal-region tiles)
            if (k0 + KT - 1 > qw) {
#pragma unroll
                for (int j = 0; j < 4; ++j) {
                    int kcol = k0 + j * 16 + (lane & 15);
#pragma unroll
                    for (int r = 0; r < 8; ++r)
                        if (kcol > qrow0 + r) sc[j][r] = NEG_INF;
                }
            }
            // row max across 4 tiles then across 16 lanes of half-wave
            float nm[8];
#pragma unroll
            for (int r = 0; r < 8; ++r) {
                float v = fmaxf(fmaxf(sc[0][r], sc[1][r]), fmaxf(sc[2][r], sc[3][r]));
#pragma unroll
                for (int msk = 1; msk < 16; msk <<= 1)
                    v = fmaxf(v, __shfl_xor(v, msk, 32));
                nm[r] = v;
            }
            float alpha[8];
#pragma unroll
            for (int r = 0; r < 8; ++r) {
                float mo = mrow[r];
                float mn = fmaxf(mo, nm[r]);
                mrow[r] = mn;
                alpha[r] = (mo == NEG_INF) ? 0.0f : __expf(mo - mn);
            }
            // P = exp(S - m), row sums
            float rs[8];
#pragma unroll
            for (int r = 0; r < 8; ++r) rs[r] = 0.0f;
#pragma unroll
            for (int j = 0; j < 4; ++j)
#pragma unroll
                for (int r = 0; r < 8; ++r) {
                    float p = __expf(sc[j][r] - mrow[r]);
                    sc[j][r] = p;
                    rs[r] += p;
                }
#pragma unroll
            for (int r = 0; r < 8; ++r) {
#pragma unroll
                for (int msk = 1; msk < 16; msk <<= 1)
                    rs[r] += __shfl_xor(rs[r], msk, 32);
                lrow[r] = lrow[r] * alpha[r] + rs[r];
            }
            // rescale output accumulators
#pragma unroll
            for (int dt = 0; dt < 4; ++dt)
#pragma unroll
                for (int r = 0; r < 8; ++r) oacc[dt][r] *= alpha[r];

            // C-layout -> A-layout via wave-private LDS scratch
            half_t* pw = &sP[wave * 16 * LDK];
#pragma unroll
            for (int j = 0; j < 4; ++j)
#pragma unroll
                for (int r = 0; r < 8; ++r)
                    pw[(rowHi + r) * LDK + j * 16 + (lane & 15)] = (half_t)sc[j][r];
            asm volatile("s_wait_dscnt 0" ::: "memory");   // wave-local RAW

            // O += P V   (keys = 2 k-steps of 32)
#pragma unroll
            for (int s = 0; s < 2; ++s) {
                v16h pf = load_frag_a(pw + s * 32, LDK, lane);
#pragma unroll
                for (int dt = 0; dt < 4; ++dt) {
                    v16h vf = load_frag_b(&sV[(dt * 16) * LDK + s * 32], LDK, lane);
                    oacc[dt] = wmma_f16(pf, vf, oacc[dt]);
                }
            }
        }
        __syncthreads();
    }

    // normalize and store to [B,T,C] f16 (C = H*D packed)
    const int b = bh >> 4, h = bh & 15;
#pragma unroll
    for (int dt = 0; dt < 4; ++dt) {
        int d = h * Dh + dt * 16 + (lane & 15);
#pragma unroll
        for (int r = 0; r < 8; ++r) {
            float inv = 1.0f / lrow[r];
            Og[(size_t)(b * Tseq + qrow0 + r) * Cdim + d] =
                (half_t)(oacc[dt][r] * inv);
        }
    }
}

// ---------------------------------------------------------------------------
extern "C" void kernel_launch(void* const* d_in, const int* in_sizes, int n_in,
                              void* d_out, int out_size, void* d_ws, size_t ws_size,
                              hipStream_t stream) {
    const float* x      = (const float*)d_in[0];
    const float* w_qkv  = (const float*)d_in[1];
    const float* w_proj = (const float*)d_in[2];
    const float* b_proj = (const float*)d_in[3];
    float* out = (float*)d_out;

    char* ws = (char*)d_ws;
    size_t off = 0;
    auto carve = [&](size_t nelem) -> half_t* {
        half_t* p = (half_t*)(ws + off);
        off += ((nelem * sizeof(half_t)) + 255) & ~(size_t)255;
        return p;
    };
    half_t* xh  = carve((size_t)Mrows * Cdim);       // x in f16
    half_t* wqh = carve((size_t)Cdim * 3 * Cdim);    // w_qkv f16
    half_t* wph = carve((size_t)Cdim * Cdim);        // w_proj f16
    half_t* Qh  = carve((size_t)Mrows * Cdim);       // [B,H,T,D]
    half_t* Kh  = carve((size_t)Mrows * Cdim);
    half_t* Vh  = carve((size_t)Mrows * Cdim);
    half_t* Ah  = carve((size_t)Mrows * Cdim);       // attn out [B,T,C]

    {
        int n4 = Mrows * Cdim / 4;
        cvt_f32_f16<<<dim3((n4 + 255) / 256), 256, 0, stream>>>(x, xh, n4);
        n4 = Cdim * 3 * Cdim / 4;
        cvt_f32_f16<<<dim3((n4 + 255) / 256), 256, 0, stream>>>(w_qkv, wqh, n4);
        n4 = Cdim * Cdim / 4;
        cvt_f32_f16<<<dim3((n4 + 255) / 256), 256, 0, stream>>>(w_proj, wph, n4);
    }

    // QKV projection: [4096,1024] x [1024,3072] -> scatter to Q/K/V
    gemm_wmma<0><<<dim3(3 * Cdim / 128, Mrows / 128), 256, 0, stream>>>(
        xh, wqh, Cdim, 3 * Cdim, Qh, Kh, Vh, nullptr, nullptr);

    // causal flash attention
    attn_flash<<<dim3(Tseq / 128, Bsz * Hn), 256, 0, stream>>>(Qh, Kh, Vh, Ah);

    // output projection + bias: [4096,1024] x [1024,1024] -> fp32 out
    gemm_wmma<1><<<dim3(Cdim / 128, Mrows / 128), 256, 0, stream>>>(
        Ah, wph, Cdim, Cdim, nullptr, nullptr, nullptr, out, b_proj);
}
